// Sampler_32452772889203
// MI455X (gfx1250) — matrix operand order repaired and checked by reading, not compiled, not run on hardware
//
#include <hip/hip_runtime.h>
#include <hip/hip_bf16.h>

// ---------------------------------------------------------------------------
// Sampler: logits GEMM (WMMA f16, fp32 accum) -> softmax stats + top-64
// selection -> top-p/top-k truncated multinomial sample.
// B=64, S=1, D=2048, V=128000
// ---------------------------------------------------------------------------

#define BATCH 64
#define SEQ   1
#define DIM   2048
#define VOCAB 128000

typedef __attribute__((ext_vector_type(16))) _Float16 v16h;
typedef __attribute__((ext_vector_type(8)))  float    v8f;
typedef __attribute__((ext_vector_type(4)))  float    flt4;

// ---------------- Kernel 1: logits = (x[:,pos,:] @ emb^T + bias) / temp -----
// 8 waves / block, each wave owns NTPW*16 vocab columns, all 64 batch rows.
// Embedding is streamed exactly once (non-temporal); x stays L2-hot, and each
// A fragment is reused across NTPW=4 B tiles to keep L2 read demand ~1 GB.

#define WAVES1 8
#define NTPW   4
#define TPB1   (WAVES1 * 32)
#define WG_N   (WAVES1 * NTPW * 16)   // 512 vocab columns per workgroup -> 250 WGs

__device__ __forceinline__ v16h cvt16(flt4 a, flt4 b, flt4 c, flt4 d) {
  v16h r;
#pragma unroll
  for (int i = 0; i < 4; ++i) {
    r[i]      = (_Float16)a[i];
    r[4 + i]  = (_Float16)b[i];
    r[8 + i]  = (_Float16)c[i];
    r[12 + i] = (_Float16)d[i];
  }
  return r;
}

__global__ __launch_bounds__(TPB1)
void logits_wmma_kernel(const float* __restrict__ emb,
                        const float* __restrict__ x,
                        const long long* __restrict__ output_pos,
                        const float* __restrict__ temperature,
                        const float* __restrict__ bias,
                        float* __restrict__ logits) {
  const int tid  = threadIdx.x;
  const int wave = tid >> 5;
  const int lane = tid & 31;
  const int half = lane >> 4;    // 0: lanes 0-15, 1: lanes 16-31
  const int l16  = lane & 15;

  const long long p0 = output_pos[0];
  const float* xs = x + p0 * (long long)DIM;   // row m lives at xs + m*SEQ*DIM

  const int vwave = blockIdx.x * WG_N + wave * (NTPW * 16);

  v8f acc[4][NTPW];
  const v8f vzero = {};
#pragma unroll
  for (int mt = 0; mt < 4; ++mt)
#pragma unroll
    for (int nt = 0; nt < NTPW; ++nt) acc[mt][nt] = vzero;

  for (int k0 = 0; k0 < DIM; k0 += 32) {
    // ---- A fragments (activations), ISA 16-bit A 16x32 layout ----
    // lane<16: elems 0..7 = K k0+0..7,  elems 8..15 = K k0+16..23  (row M=l16)
    // lane>=16: elems 0..7 = K k0+8..15, elems 8..15 = K k0+24..31
    v16h afrag[4];
#pragma unroll
    for (int mt = 0; mt < 4; ++mt) {
      const float* ap = xs + (long long)(mt * 16 + l16) * (SEQ * DIM) + k0 + half * 8;
      flt4 a0 = *(const flt4*)(ap + 0);
      flt4 a1 = *(const flt4*)(ap + 4);
      flt4 a2 = *(const flt4*)(ap + 16);
      flt4 a3 = *(const flt4*)(ap + 20);
      afrag[mt] = cvt16(a0, a1, a2, a3);
    }
    // ---- B fragments (embedding rows as columns), 16-bit B 32x16 layout ----
    // lane: column N = l16, elems 0..7 = K k0+half*8 .. +7, elems 8..15 = K k0+16+half*8 .. +7
#pragma unroll
    for (int nt = 0; nt < NTPW; ++nt) {
      const float* bp = emb + (long long)(vwave + nt * 16 + l16) * DIM + k0 + half * 8;
      flt4 b0 = __builtin_nontemporal_load((const flt4*)(bp + 0));
      flt4 b1 = __builtin_nontemporal_load((const flt4*)(bp + 4));
      flt4 b2 = __builtin_nontemporal_load((const flt4*)(bp + 16));
      flt4 b3 = __builtin_nontemporal_load((const flt4*)(bp + 20));
      v16h bfrag = cvt16(b0, b1, b2, b3);
#pragma unroll
      for (int mt = 0; mt < 4; ++mt) {
        acc[mt][nt] = __builtin_amdgcn_wmma_f32_16x16x32_f16(
            /*neg_a=*/false, afrag[mt], /*neg_b=*/false, bfrag,
            /*c_mod=*/(short)0, acc[mt][nt],
            /*reuse_a=*/false, /*reuse_b=*/false);
      }
    }
  }

  // ---- epilogue: + bias, * (1/temperature), scatter to logits[m][v] ----
  // C/D layout: value at (vgpr r, lane) is row M = r + 8*half, col N = l16.
  // Hoist one v_rcp per (mt,r) row instead of a full divide per element.
  float invT[4][8];
#pragma unroll
  for (int mt = 0; mt < 4; ++mt)
#pragma unroll
    for (int r = 0; r < 8; ++r)
      invT[mt][r] = __builtin_amdgcn_rcpf(temperature[mt * 16 + r + 8 * half]);

#pragma unroll
  for (int nt = 0; nt < NTPW; ++nt) {
    const int v = vwave + nt * 16 + l16;
    const float bv = bias[v];
#pragma unroll
    for (int mt = 0; mt < 4; ++mt) {
#pragma unroll
      for (int r = 0; r < 8; ++r) {
        const int m = mt * 16 + r + 8 * half;
        logits[(long long)m * VOCAB + v] = (acc[mt][nt][r] + bv) * invT[mt][r];
      }
    }
  }
}

// ---------------- Kernel 2: per-row max, sum-exp, top-64 selection ----------

#define TPB2  1024
#define CHUNK (VOCAB / TPB2)   // 125 contiguous elements per thread
#define TOPN  64               // >= max(topk)=50

__device__ __forceinline__ unsigned ford(float f) {
  unsigned u = __float_as_uint(f);
  return (u & 0x80000000u) ? ~u : (u | 0x80000000u);
}
__device__ __forceinline__ float frev(unsigned o) {
  unsigned u = (o & 0x80000000u) ? (o & 0x7FFFFFFFu) : ~o;
  return __uint_as_float(u);
}

__device__ __forceinline__ void rescan(const float* __restrict__ row, int base,
                                       unsigned long long m0, unsigned long long m1,
                                       float& cv, int& ci) {
  cv = -3.4e38f; ci = -1;
  for (int j = 0; j < CHUNK; ++j) {
    bool taken = (j < 64) ? ((m0 >> j) & 1ull) : ((m1 >> (j - 64)) & 1ull);
    if (taken) continue;
    float v = row[base + j];
    if (v > cv) { cv = v; ci = base + j; }   // first-hit keeps smallest idx on tie
  }
}

__global__ __launch_bounds__(TPB2)
void topk_stats_kernel(const float* __restrict__ logits,
                       float* __restrict__ topW,
                       int* __restrict__ topIdx,
                       float* __restrict__ stats) {
  __shared__ float redf[32];
  __shared__ unsigned long long redk[32];
  __shared__ float sMax, sZ;
  __shared__ unsigned long long sWin;

  const int b = blockIdx.x;
  const int t = threadIdx.x;
  const int wave = t >> 5, lane = t & 31;
  const float* row = logits + (long long)b * VOCAB;
  const int base = t * CHUNK;

  // ---- row max ----
  float m = -3.4e38f;
  for (int j = 0; j < CHUNK; ++j) m = fmaxf(m, row[base + j]);
#pragma unroll
  for (int off = 16; off; off >>= 1) m = fmaxf(m, __shfl_xor(m, off, 32));
  if (lane == 0) redf[wave] = m;
  __syncthreads();
  if (wave == 0) {
    float v = redf[lane];
#pragma unroll
    for (int off = 16; off; off >>= 1) v = fmaxf(v, __shfl_xor(v, off, 32));
    if (lane == 0) sMax = v;
  }
  __syncthreads();
  const float rowMax = sMax;

  // ---- row sum-exp (softmax denominator Z) ----
  float s = 0.f;
  for (int j = 0; j < CHUNK; ++j) s += __expf(row[base + j] - rowMax);
#pragma unroll
  for (int off = 16; off; off >>= 1) s += __shfl_xor(s, off, 32);
  if (lane == 0) redf[wave] = s;
  __syncthreads();
  if (wave == 0) {
    float v = redf[lane];
#pragma unroll
    for (int off = 16; off; off >>= 1) v += __shfl_xor(v, off, 32);
    if (lane == 0) { sZ = v; }
  }
  __syncthreads();
  if (t == 0) { stats[b * 2 + 0] = rowMax; stats[b * 2 + 1] = sZ; }

  // ---- iterative top-64 (descending, stable) ----
  unsigned long long m0 = 0ull, m1 = 0ull;   // 125-bit taken mask
  float cv; int ci;
  rescan(row, base, m0, m1, cv, ci);

  for (int it = 0; it < TOPN; ++it) {
    unsigned long long key =
        ((unsigned long long)ford(cv) << 32) | (unsigned)(~(unsigned)ci);
#pragma unroll
    for (int off = 16; off; off >>= 1) {
      unsigned long long o = __shfl_xor(key, off, 32);
      if (o > key) key = o;
    }
    if (lane == 0) redk[wave] = key;
    __syncthreads();
    if (wave == 0) {
      unsigned long long k2 = redk[lane];
#pragma unroll
      for (int off = 16; off; off >>= 1) {
        unsigned long long o = __shfl_xor(k2, off, 32);
        if (o > k2) k2 = o;
      }
      if (lane == 0) sWin = k2;
    }
    __syncthreads();
    const unsigned long long win = sWin;
    const int wi = (int)(~(unsigned)(win & 0xFFFFFFFFull));
    if (t == 0) {
      topW[b * TOPN + it]   = __expf(frev((unsigned)(win >> 32)) - rowMax);
      topIdx[b * TOPN + it] = wi;
    }
    if (wi >= base && wi < base + CHUNK) {   // owner updates mask, refreshes cand
      const int j = wi - base;
      if (j < 64) m0 |= 1ull << j; else m1 |= 1ull << (j - 64);
      rescan(row, base, m0, m1, cv, ci);
    }
    __syncthreads();
  }
}

// ---------------- Kernel 3: top-p prefix + top-k cap, renormalize, sample ---

__global__ void sample_kernel(const float* __restrict__ topW,
                              const int* __restrict__ topIdx,
                              const float* __restrict__ stats,
                              const float* __restrict__ topp,
                              const int* __restrict__ topk,
                              float* __restrict__ out) {
  const int b = threadIdx.x;
  if (b >= BATCH) return;

  const float Z  = stats[b * 2 + 1];
  const float tp = topp[b];
  int tk = topk[b]; if (tk > TOPN) tk = TOPN; if (tk < 1) tk = 1;

  // keep sorted prefix while cumulative prob BEFORE token <= topp, and j < topk
  float cum = 0.f, kept = 0.f; int L = 0;
  for (int j = 0; j < tk; ++j) {
    if (cum > tp) break;
    const float w = topW[b * TOPN + j];
    cum  += w / Z;
    kept += w;
    L = j + 1;
  }

  // deterministic uniform (murmur-style finalize of (123, b))
  unsigned h = 123u ^ ((unsigned)b * 0x9E3779B9u);
  h ^= h >> 16; h *= 0x85EBCA6Bu; h ^= h >> 13; h *= 0xC2B2AE35u; h ^= h >> 16;
  const float u = (float)(h >> 8) * (1.0f / 16777216.0f);

  // inverse-CDF over kept (renormalization by `kept` folded into target)
  const float target = u * kept;
  float csum = 0.f;
  int pick = topIdx[b * TOPN];
  for (int j = 0; j < L; ++j) {
    csum += topW[b * TOPN + j];
    if (csum >= target) { pick = topIdx[b * TOPN + j]; break; }
  }
  out[b] = (float)pick;
}

// ---------------------------------------------------------------------------

extern "C" void kernel_launch(void* const* d_in, const int* in_sizes, int n_in,
                              void* d_out, int out_size, void* d_ws, size_t ws_size,
                              hipStream_t stream) {
  const float*     emb   = (const float*)d_in[0];       // [V, D]
  const float*     x     = (const float*)d_in[1];       // [B, S, D]
  const long long* pos   = (const long long*)d_in[2];   // [1]
  const float*     temp  = (const float*)d_in[3];       // [B]
  const float*     topp  = (const float*)d_in[4];       // [B]
  const int*       topk  = (const int*)d_in[5];         // [B]
  const float*     bias  = (const float*)d_in[6];       // [V]

  // workspace layout
  float* logits = (float*)d_ws;                                    // B*V floats (32.8 MB)
  float* topW   = logits + (size_t)BATCH * VOCAB;                  // B*TOPN
  int*   topIdx = (int*)(topW + (size_t)BATCH * TOPN);             // B*TOPN
  float* stats  = (float*)(topIdx + (size_t)BATCH * TOPN);         // B*2 (max, Z)

  logits_wmma_kernel<<<VOCAB / WG_N, TPB1, 0, stream>>>(emb, x, pos, temp, bias, logits);
  topk_stats_kernel<<<BATCH, TPB2, 0, stream>>>(logits, topW, topIdx, stats);
  sample_kernel<<<1, BATCH, 0, stream>>>(topW, topIdx, stats, topp, topk, (float*)d_out);
}
